// TransformerLayer_4629974745875
// MI455X (gfx1250) — matrix operand order repaired and checked by reading, not compiled, not run on hardware
//
#include <hip/hip_runtime.h>

// ---------------------------------------------------------------------------
// MI455X (gfx1250, wave32) transformer layer.
// All GEMMs in bf16 via v_wmma_f32_16x16x32_bf16; softmax/LN in fp32.
// GEMM A-tiles streamed via CDNA5 async global->LDS DMA (ASYNCcnt) when the
// toolchain exposes the builtin; register-staged fallback otherwise.
// ---------------------------------------------------------------------------

typedef __attribute__((ext_vector_type(16))) __bf16 v16bf;
typedef __attribute__((ext_vector_type(8)))  float  v8f;
typedef __attribute__((ext_vector_type(4)))  int    v4i;

typedef unsigned short bfbits;   // bf16 bit pattern storage

#if defined(__gfx1250__) && __has_builtin(__builtin_amdgcn_global_load_async_to_lds_b128)
#define HAS_ASYNC_LDS 1
#else
#define HAS_ASYNC_LDS 0
#endif

#if HAS_ASYNC_LDS
// builtin prototype: (v4i AS1*, v4i AS3*, imm int offset, imm int cpol)
typedef __attribute__((address_space(1))) v4i* gv4i_ptr;
typedef __attribute__((address_space(3))) v4i* lv4i_ptr;
#endif

union FragB16 {
  v16bf v;
  unsigned int d[8];
  unsigned short u[16];
};

__device__ __forceinline__ bfbits f2bf(float f) {
  unsigned int u = __float_as_uint(f);
  u += 0x7FFFu + ((u >> 16) & 1u);   // round to nearest even
  return (bfbits)(u >> 16);
}

__device__ __forceinline__ v8f wmma_bf16(v16bf a, v16bf b, v8f c) {
  // (neg_a, A, neg_b, B, c_mod, C, reuse_a, reuse_b)
  return __builtin_amdgcn_wmma_f32_16x16x32_bf16(false, a, false, b,
                                                 (short)0, c, false, false);
}

// A-matrix fragment (16xK tile, element (m,k) at base[m*ld + k]).
// A 16x32 layout: lane = m + 16*(khalf); lane half h holds k in [16h,16h+16),
// vgpr i packs k = 16h+2i, 16h+2i+1 (matches B fragments).
__device__ __forceinline__ v16bf frag_a(const bfbits* base, int ld, int lane) {
  const bfbits* row = base + (size_t)(lane & 15) * ld;
  const int kb = (lane >> 4) << 4;   // 0 or 16
  FragB16 f;
#pragma unroll
  for (int i = 0; i < 8; ++i) {
    const int k = kb + (i << 1);
    f.d[i] = *(const unsigned int*)(row + k);
  }
  return f.v;
}

// B-matrix fragment, column for this lane is CONTIGUOUS in memory
// (colbase[k] == element (k, n_of_this_lane)).  Lane n = lane&15,
// k-half = (lane>>4)*16, vgpr v packs k = kh+2v, kh+2v+1.
__device__ __forceinline__ v16bf frag_b_cont(const bfbits* colbase, int lane) {
  const int kb = (lane >> 4) << 4;   // 0 or 16
  FragB16 f;
#pragma unroll
  for (int i = 0; i < 8; ++i)
    f.d[i] = *(const unsigned int*)(colbase + kb + (i << 1));
  return f.v;
}

// B-matrix fragment from row-major K x N storage: element (k,n) = base[k*ld+n].
__device__ __forceinline__ v16bf frag_b_rowmaj(const bfbits* base, int ld, int lane) {
  const int n  = lane & 15;
  const int kb = (lane >> 4) << 4;
  FragB16 f;
#pragma unroll
  for (int i = 0; i < 8; ++i) {
    const int k = kb + (i << 1);
    f.u[2 * i]     = base[(size_t)k * ld + n];
    f.u[2 * i + 1] = base[(size_t)(k + 1) * ld + n];
  }
  return f.v;
}

__device__ __forceinline__ float rmax16(float v) {
#pragma unroll
  for (int off = 8; off; off >>= 1) v = fmaxf(v, __shfl_xor(v, off, 32));
  return v;
}
__device__ __forceinline__ float rsum16(float v) {
#pragma unroll
  for (int off = 8; off; off >>= 1) v += __shfl_xor(v, off, 32);
  return v;
}

// ---------------------------------------------------------------------------
// bf16 GEMM: C[M,N] = A[M,K] * B[K,N] (+ epilogue).  M,N mult of 128, K of 32.
// EPI 0: QKV scatter -> q/k/v bf16 [B,H,S,64]      (bias f32)
// EPI 1: outF = acc + bias + resid   (fp32)
// EPI 2: outH = relu(acc + bias)     (bf16)
// A tile: double-buffered LDS, filled by async global->LDS DMA (ASYNCcnt).
// B tile: transposed in LDS (Bs[n][k]) so fragment reads are contiguous b32.
// ---------------------------------------------------------------------------
template <int EPI>
__global__ __launch_bounds__(256) void gemm_bf16(
    const bfbits* __restrict__ A, const bfbits* __restrict__ B,
    int M, int N, int K,
    const float* __restrict__ bias, const float* __restrict__ resid,
    float* __restrict__ outF, bfbits* __restrict__ outH,
    bfbits* __restrict__ qo, bfbits* __restrict__ ko, bfbits* __restrict__ vo) {
  constexpr int BM = 128, BN = 128, BK = 32;
  __shared__ bfbits As[2][BM][BK];  // row-major tile, double buffered
  __shared__ bfbits Bs[BN][BK];     // TRANSPOSED tile: Bs[n][k]

  const int tid  = threadIdx.x;
  const int lane = tid & 31;
  const int wid  = tid >> 5;
  const int wm   = wid & 1;         // 2 wave-rows  (64 rows each)
  const int wn   = wid >> 1;        // 4 wave-cols  (32 cols each)
  const int bm   = blockIdx.y * BM;
  const int bn   = blockIdx.x * BN;

  v8f acc[4][2];
#pragma unroll
  for (int mt = 0; mt < 4; ++mt)
#pragma unroll
    for (int nt = 0; nt < 2; ++nt) acc[mt][nt] = (v8f){};

  // per-thread tile chunks (2 x 16B for A, 2 x 16B for B)
  int arow[2], aseg[2], brow[2], bseg[2];
#pragma unroll
  for (int i = 0; i < 2; ++i) {
    const int c = tid + i * 256;          // 512 chunks per tile
    arow[i] = c >> 2;  aseg[i] = (c & 3) * 8;    // A: 128 rows x 4 chunks
    brow[i] = c >> 4;  bseg[i] = (c & 15) * 8;   // B: 32 rows x 16 chunks
  }

#if HAS_ASYNC_LDS
  auto a_fill = [&](int kt, int buf) {   // async DMA global -> LDS
#pragma unroll
    for (int i = 0; i < 2; ++i)
      __builtin_amdgcn_global_load_async_to_lds_b128(
          (gv4i_ptr)(v4i*)(A + (size_t)(bm + arow[i]) * K + kt + aseg[i]),
          (lv4i_ptr)(v4i*)&As[buf][arow[i]][aseg[i]], 0, 0);
  };
  auto a_wait = [&]() { asm volatile("s_wait_asynccnt 0" ::: "memory"); };
#else
  uint4 ra[2];
  auto a_gload = [&](int kt) {
#pragma unroll
    for (int i = 0; i < 2; ++i)
      ra[i] = *(const uint4*)(A + (size_t)(bm + arow[i]) * K + kt + aseg[i]);
  };
  auto a_sstore = [&](int buf) {
#pragma unroll
    for (int i = 0; i < 2; ++i)
      *(uint4*)&As[buf][arow[i]][aseg[i]] = ra[i];
  };
#endif

  uint4 rb[2];
  auto b_gload = [&](int kt) {
#pragma unroll
    for (int i = 0; i < 2; ++i)
      rb[i] = *(const uint4*)(B + (size_t)(kt + brow[i]) * N + bn + bseg[i]);
  };
  auto b_sstore = [&]() {
#pragma unroll
    for (int i = 0; i < 2; ++i) {
      const unsigned short* src = (const unsigned short*)&rb[i];
#pragma unroll
      for (int j = 0; j < 8; ++j) Bs[bseg[i] + j][brow[i]] = src[j];
    }
  };

  // prologue: tile 0 into buffer 0
#if HAS_ASYNC_LDS
  a_fill(0, 0);
#else
  a_gload(0);
#endif
  b_gload(0);
  b_sstore();
#if HAS_ASYNC_LDS
  a_wait();
#else
  a_sstore(0);
#endif
  __syncthreads();

  const int nk = K / BK;
  for (int t = 0; t < nk; ++t) {
    const int buf = t & 1, nbuf = buf ^ 1;
    if (t + 1 < nk) {                       // prefetch overlaps WMMA below
#if HAS_ASYNC_LDS
      a_fill((t + 1) * BK, nbuf);           // DMA into the other buffer
#else
      a_gload((t + 1) * BK);
#endif
      b_gload((t + 1) * BK);
    }

    v16bf af[4];
#pragma unroll
    for (int mt = 0; mt < 4; ++mt)
      af[mt] = frag_a(&As[buf][wm * 64 + mt * 16][0], BK, lane);
    v16bf bfr[2];
#pragma unroll
    for (int nt = 0; nt < 2; ++nt)
      bfr[nt] = frag_b_cont(&Bs[wn * 32 + nt * 16 + (lane & 15)][0], lane);

#pragma unroll
    for (int mt = 0; mt < 4; ++mt)
#pragma unroll
      for (int nt = 0; nt < 2; ++nt)
        acc[mt][nt] = wmma_bf16(af[mt], bfr[nt], acc[mt][nt]);

    __syncthreads();
    if (t + 1 < nk) {
      b_sstore();
#if HAS_ASYNC_LDS
      a_wait();                             // A(t+1) landed in LDS
#else
      a_sstore(nbuf);
#endif
    }
    __syncthreads();
  }

  // epilogue: C layout — lane n = lane&15, rows mbase..mbase+7 per vgpr r
  const int nloc  = lane & 15;
  const int moff  = (lane >> 4) << 3;
#pragma unroll
  for (int mt = 0; mt < 4; ++mt) {
#pragma unroll
    for (int nt = 0; nt < 2; ++nt) {
      const int gn = bn + wn * 32 + nt * 16 + nloc;
      const int mb = bm + wm * 64 + mt * 16 + moff;
      const float bv = bias[gn];
#pragma unroll
      for (int r = 0; r < 8; ++r) {
        const int gm = mb + r;
        const float val = acc[mt][nt][r] + bv;
        if (EPI == 1) {
          outF[(size_t)gm * N + gn] = val + resid[(size_t)gm * N + gn];
        } else if (EPI == 2) {
          outH[(size_t)gm * N + gn] = f2bf(fmaxf(val, 0.f));
        } else {  // QKV scatter: gm -> (b,s), gn -> (which, head, d)
          const int bb = gm >> 10, ss = gm & 1023;
          const int which = gn / 768;
          const int w = gn - which * 768;
          const int hh = w >> 6, dd = w & 63;
          bfbits* dst = (which == 0) ? qo : (which == 1) ? ko : vo;
          dst[((((size_t)bb * 12 + hh) << 10) + ss) * 64 + dd] = f2bf(val);
        }
      }
    }
  }
}

// ---------------------------------------------------------------------------
// Causal flash attention. Q/K/V bf16 [B*H, S, 64]. One wave = 16 query rows,
// block = 128 rows of one (b,h).  Key blocks of 32.
// ---------------------------------------------------------------------------
__global__ __launch_bounds__(256) void attn_kernel(
    const bfbits* __restrict__ Q, const bfbits* __restrict__ Km,
    const bfbits* __restrict__ V, bfbits* __restrict__ ctx) {
  __shared__ bfbits Ps[8][16][32];   // per-wave P staging (C->A layout swap)

  const int lane = threadIdx.x & 31;
  const int wid  = threadIdx.x >> 5;
  const int bh   = blockIdx.y;          // 0..95
  const int bb   = bh / 12, hh = bh % 12;
  const int q0   = blockIdx.x * 128 + wid * 16;

  const bfbits* Qp = Q + ((size_t)bh << 16);   // 1024*64 elems per head
  const bfbits* Kp = Km + ((size_t)bh << 16);
  const bfbits* Vp = V + ((size_t)bh << 16);

  const v16bf qf0 = frag_a(Qp + (size_t)q0 * 64, 64, lane);
  const v16bf qf1 = frag_a(Qp + (size_t)q0 * 64 + 32, 64, lane);

  v8f o[4];
#pragma unroll
  for (int c = 0; c < 4; ++c) o[c] = (v8f){};
  float mrow[8], lrow[8];
#pragma unroll
  for (int r = 0; r < 8; ++r) { mrow[r] = -3.0e38f; lrow[r] = 0.f; }

  const int nloc = lane & 15;
  const int moff = (lane >> 4) << 3;
  const int kend = q0 + 16;             // keys < kend can be unmasked

  for (int kb = 0; kb < kend; kb += 32) {
    // ---- scores: S[16x32] = Q (16x64) * K^T, two 16x16 C tiles ----
    v8f s0 = (v8f){}, s1 = (v8f){};
    {
      const bfbits* col0 = Kp + (size_t)(kb + nloc) * 64;        // keys kb..+15
      const bfbits* col1 = Kp + (size_t)(kb + 16 + nloc) * 64;   // keys +16..+31
      s0 = wmma_bf16(qf0, frag_b_cont(col0, lane), s0);
      s0 = wmma_bf16(qf1, frag_b_cont(col0 + 32, lane), s0);
      s1 = wmma_bf16(qf0, frag_b_cont(col1, lane), s1);
      s1 = wmma_bf16(qf1, frag_b_cont(col1 + 32, lane), s1);
    }

    // ---- scale + causal mask (match reference: /8 then -1e9) ----
    const bool needmask = (kb + 31) > q0;
#pragma unroll
    for (int r = 0; r < 8; ++r) {
      float a0 = s0[r] * 0.125f;
      float a1 = s1[r] * 0.125f;
      if (needmask) {
        const int qrow = q0 + moff + r;
        if (kb + nloc > qrow) a0 = -1e9f;
        if (kb + 16 + nloc > qrow) a1 = -1e9f;
      }
      s0[r] = a0; s1[r] = a1;
    }

    // ---- online softmax ----
    float tmax[8];
#pragma unroll
    for (int r = 0; r < 8; ++r) tmax[r] = rmax16(fmaxf(s0[r], s1[r]));
#pragma unroll
    for (int r = 0; r < 8; ++r) {
      const float mnew  = fmaxf(mrow[r], tmax[r]);
      const float alpha = __expf(mrow[r] - mnew);
      mrow[r] = mnew;
      lrow[r] *= alpha;
#pragma unroll
      for (int c = 0; c < 4; ++c) o[c][r] *= alpha;
      s0[r] = __expf(s0[r] - mnew);
      s1[r] = __expf(s1[r] - mnew);
    }
#pragma unroll
    for (int r = 0; r < 8; ++r) lrow[r] += rsum16(s0[r] + s1[r]);

    // ---- P (C layout, fp32) -> LDS bf16 -> A fragment ----
#pragma unroll
    for (int r = 0; r < 8; ++r) {
      Ps[wid][moff + r][nloc]      = f2bf(s0[r]);
      Ps[wid][moff + r][16 + nloc] = f2bf(s1[r]);
    }
    asm volatile("s_wait_dscnt 0" ::: "memory");
    const v16bf pf = frag_a(&Ps[wid][0][0], 32, lane);

    // ---- O += P (16x32) * V (32x64) ----
#pragma unroll
    for (int c = 0; c < 4; ++c) {
      const v16bf vf = frag_b_rowmaj(Vp + (size_t)kb * 64 + c * 16, 64, lane);
      o[c] = wmma_bf16(pf, vf, o[c]);
    }
  }

  // ---- normalize, store ctx as bf16 [B,S,768] (head-interleaved) ----
#pragma unroll
  for (int c = 0; c < 4; ++c) {
#pragma unroll
    for (int r = 0; r < 8; ++r) {
      const float val = o[c][r] / lrow[r];
      const int srow = q0 + moff + r;
      ctx[((size_t)bb * 1024 + srow) * 768 + hh * 64 + c * 16 + nloc] = f2bf(val);
    }
  }
}

// ---------------------------------------------------------------------------
// LayerNorm over 768, one block per row.
// ---------------------------------------------------------------------------
template <bool WB>
__global__ __launch_bounds__(256) void layernorm_kernel(
    const float* __restrict__ in, const float* __restrict__ g,
    const float* __restrict__ be, float* __restrict__ outF,
    bfbits* __restrict__ outB) {
  __shared__ float red[256];
  const int tid = threadIdx.x;
  const int row = blockIdx.x;
  const float* x = in + (size_t)row * 768;

  float s = 0.f;
  for (int i = tid; i < 768; i += 256) s += x[i];
  red[tid] = s; __syncthreads();
  for (int st = 128; st > 0; st >>= 1) {
    if (tid < st) red[tid] += red[tid + st];
    __syncthreads();
  }
  const float mu = red[0] * (1.f / 768.f);
  __syncthreads();

  float v = 0.f;
  for (int i = tid; i < 768; i += 256) { const float d = x[i] - mu; v += d * d; }
  red[tid] = v; __syncthreads();
  for (int st = 128; st > 0; st >>= 1) {
    if (tid < st) red[tid] += red[tid + st];
    __syncthreads();
  }
  const float inv = rsqrtf(red[0] * (1.f / 768.f) + 1e-5f);

  for (int i = tid; i < 768; i += 256) {
    const float y = (x[i] - mu) * inv * g[i] + be[i];
    outF[(size_t)row * 768 + i] = y;
    if (WB) outB[(size_t)row * 768 + i] = f2bf(y);
  }
}

// ---------------------------------------------------------------------------
// Conversions / packing
// ---------------------------------------------------------------------------
__global__ void cvt_f32_bf16(const float* __restrict__ in,
                             bfbits* __restrict__ out, int n) {
  for (int i = blockIdx.x * blockDim.x + threadIdx.x; i < n;
       i += gridDim.x * blockDim.x)
    out[i] = f2bf(in[i]);
}

__global__ void pack_qkv(const float* __restrict__ Wq, const float* __restrict__ Wk,
                         const float* __restrict__ Wv, bfbits* __restrict__ W,
                         const float* __restrict__ bq, const float* __restrict__ bk,
                         const float* __restrict__ bv, float* __restrict__ bcat) {
  const int i = blockIdx.x * blockDim.x + threadIdx.x;
  if (i < 768 * 2304) {
    const int k = i / 2304, j = i - (i / 2304) * 2304;
    const float* src = (j < 768) ? Wq : (j < 1536) ? Wk : Wv;
    const int jj = (j < 768) ? j : (j < 1536) ? j - 768 : j - 1536;
    W[i] = f2bf(src[(size_t)k * 768 + jj]);
  }
  if (i < 2304)
    bcat[i] = (i < 768) ? bq[i] : (i < 1536) ? bk[i - 768] : bv[i - 1536];
}

// ---------------------------------------------------------------------------
extern "C" void kernel_launch(void* const* d_in, const int* in_sizes, int n_in,
                              void* d_out, int out_size, void* d_ws,
                              size_t ws_size, hipStream_t stream) {
  (void)in_sizes; (void)n_in; (void)out_size; (void)ws_size;
  const float* x   = (const float*)d_in[0];
  const float* Wq  = (const float*)d_in[2];
  const float* bq  = (const float*)d_in[3];
  const float* Wk  = (const float*)d_in[4];
  const float* bk  = (const float*)d_in[5];
  const float* Wv  = (const float*)d_in[6];
  const float* bv  = (const float*)d_in[7];
  const float* Wo  = (const float*)d_in[8];
  const float* bo  = (const float*)d_in[9];
  const float* W1  = (const float*)d_in[10];
  const float* b1  = (const float*)d_in[11];
  const float* W2  = (const float*)d_in[12];
  const float* b2  = (const float*)d_in[13];
  const float* g1  = (const float*)d_in[14];
  const float* be1 = (const float*)d_in[15];
  const float* g2  = (const float*)d_in[16];
  const float* be2 = (const float*)d_in[17];

  char* p = (char*)d_ws;
  auto alloc = [&](size_t bytes) {
    char* r = p;
    p += (bytes + 255) & ~(size_t)255;
    return r;
  };
  const size_t ACT = (size_t)8192 * 768;     // tokens * d_model

  bfbits* Xb    = (bfbits*)alloc(ACT * 2);             // also reused as ctx
  bfbits* Wqkvb = (bfbits*)alloc((size_t)768 * 2304 * 2);
  float*  bqkv  = (float*)alloc(2304 * 4);
  bfbits* Wob   = (bfbits*)alloc((size_t)768 * 768 * 2);
  bfbits* W1b   = (bfbits*)alloc((size_t)768 * 3072 * 2);
  bfbits* W2b   = (bfbits*)alloc((size_t)3072 * 768 * 2);
  bfbits* Qb    = (bfbits*)alloc(ACT * 2);
  bfbits* Kb    = (bfbits*)alloc(ACT * 2);
  bfbits* Vb    = (bfbits*)alloc(ACT * 2);
  float*  t1    = (float*)alloc(ACT * 4);              // also reused as t2
  float*  h     = (float*)alloc(ACT * 4);
  bfbits* hb    = (bfbits*)alloc(ACT * 2);
  bfbits* ffb   = (bfbits*)alloc((size_t)8192 * 3072 * 2);

  bfbits* ctxb = Xb;   // Xb dead after QKV GEMM
  float*  t2   = t1;   // t1 dead after LN1

  const dim3 blk(256);

  // 1. precision conversion / packing
  cvt_f32_bf16<<<2048, blk, 0, stream>>>(x, Xb, (int)ACT);
  pack_qkv<<<(768 * 2304 + 255) / 256, blk, 0, stream>>>(Wq, Wk, Wv, Wqkvb,
                                                         bq, bk, bv, bqkv);
  cvt_f32_bf16<<<1024, blk, 0, stream>>>(Wo, Wob, 768 * 768);
  cvt_f32_bf16<<<2048, blk, 0, stream>>>(W1, W1b, 768 * 3072);
  cvt_f32_bf16<<<2048, blk, 0, stream>>>(W2, W2b, 3072 * 768);

  // 2. fused QKV projection (scatter into [B,H,S,64] bf16)
  gemm_bf16<0><<<dim3(2304 / 128, 8192 / 128), blk, 0, stream>>>(
      Xb, Wqkvb, 8192, 2304, 768, bqkv, nullptr, nullptr, nullptr, Qb, Kb, Vb);

  // 3. causal flash attention -> ctx bf16 [B,S,768]
  attn_kernel<<<dim3(8, 96), blk, 0, stream>>>(Qb, Kb, Vb, ctxb);

  // 4. output projection + residual -> t1 (fp32)
  gemm_bf16<1><<<dim3(768 / 128, 8192 / 128), blk, 0, stream>>>(
      ctxb, Wob, 8192, 768, 768, bo, x, t1, nullptr, nullptr, nullptr, nullptr);

  // 5. LayerNorm1 -> h (fp32) + hb (bf16)
  layernorm_kernel<true><<<8192, blk, 0, stream>>>(t1, g1, be1, h, hb);

  // 6. FFN up + ReLU -> ffb (bf16)
  gemm_bf16<2><<<dim3(3072 / 128, 8192 / 128), blk, 0, stream>>>(
      hb, W1b, 8192, 3072, 768, b1, nullptr, nullptr, ffb, nullptr, nullptr,
      nullptr);

  // 7. FFN down + residual(h) -> t2 (fp32)
  gemm_bf16<1><<<dim3(768 / 128, 8192 / 128), blk, 0, stream>>>(
      ffb, W2b, 8192, 768, 3072, b2, h, t2, nullptr, nullptr, nullptr, nullptr);

  // 8. LayerNorm2 -> d_out (fp32)
  layernorm_kernel<false><<<8192, blk, 0, stream>>>(t2, g2, be2, (float*)d_out,
                                                    nullptr);
}
